// MLP_60215441490176
// MI455X (gfx1250) — compile-verified, matrix operand order checked by reading, not run on hardware
//
#include <hip/hip_runtime.h>
#include <hip/hip_bf16.h>
#include <math.h>

// ---------------------------------------------------------------------------
// Model constants
// ---------------------------------------------------------------------------
#define BSZ   2048
#define EDIM  128
#define HDIM  1024
#define VOCAB 100000
#define DIN   2048          // 16 * EDIM
#define EPSV  1e-5f

typedef __bf16 bf16;
typedef __attribute__((ext_vector_type(16))) __bf16 v16bf;
typedef __attribute__((ext_vector_type(8)))  __bf16 v8bf;
typedef __attribute__((ext_vector_type(8)))  float  v8f;
typedef __attribute__((ext_vector_type(4)))  int    v4i;

// ---------------------------------------------------------------------------
// CDNA5 async global->LDS (guarded; falls back to plain b128 loads)
// ---------------------------------------------------------------------------
#if defined(__has_builtin)
#  if __has_builtin(__builtin_amdgcn_global_load_async_to_lds_b128)
#    define ASYNC_LDS 1
#  endif
#  if __has_builtin(__builtin_amdgcn_s_wait_asynccnt)
#    define HAVE_WAIT_ASYNC 1
#  endif
#endif

__device__ __forceinline__ bf16 to_bf16(float f) {
  union { float f; unsigned u; } a; a.f = f;
  unsigned r = a.u + 0x7FFFu + ((a.u >> 16) & 1u);   // round-to-nearest-even
  unsigned short h = (unsigned short)(r >> 16);
  return __builtin_bit_cast(bf16, h);
}

#if ASYNC_LDS
// Signature per hipcc diagnostic: arg0 = AS(1) v4i*, arg1 = AS(3) v4i*,
// then imm offset + imm cpol.
typedef __attribute__((address_space(1))) v4i gv4i;
typedef __attribute__((address_space(3))) v4i lv4i;

__device__ __forceinline__ void async_copy16(const bf16* g, bf16* l) {
  __builtin_amdgcn_global_load_async_to_lds_b128(
      (gv4i*)(__attribute__((address_space(1))) void*)(void*)(size_t)g,
      (lv4i*)(__attribute__((address_space(3))) void*)l,
      0, 0);
}
__device__ __forceinline__ void wait_async0() {
#if HAVE_WAIT_ASYNC
  __builtin_amdgcn_s_wait_asynccnt(0);
#else
  asm volatile("s_wait_asynccnt 0" ::: "memory");
#endif
}
__device__ __forceinline__ void wait_async4() {
#if HAVE_WAIT_ASYNC
  __builtin_amdgcn_s_wait_asynccnt(4);
#else
  asm volatile("s_wait_asynccnt 4" ::: "memory");
#endif
}
#endif

// ---------------------------------------------------------------------------
// bf16 TN GEMM:  C[M,N] = A[M,K] * Bw[N,K]^T (+ bias[N])
// 256 threads = 8 waves; block tile 128x128, K-step 32, double-buffered LDS:
// iteration t issues tile t+1 via the async engine (4 in-flight b128 copies
// per thread), waits ASYNCcnt<=4 so tile t's copies (in-order) are complete,
// and overlaps tile t+1's global traffic with tile t's 8 WMMAs per wave.
// Wave (wm,wn) owns 64x32 -> 4x2 grid of 16x16 v_wmma_f32_16x16x32_bf16.
// ---------------------------------------------------------------------------
#define BK   32
#define LDT  40            // LDS row stride (elements) to stagger banks
#define TSZ  (128 * LDT)   // one LDS tile

__device__ __forceinline__ v8f wmma_bf16(v16bf a, v16bf b, v8f c) {
  return __builtin_amdgcn_wmma_f32_16x16x32_bf16(
      false, a, false, b, (short)0, c, false, false);
}

// ISA 16-bit A 16x32 layout: lanes 0-15 row m hold K 0..7 (halves 0-7) and
// K 16..23 (halves 8-15); lanes 16-31 hold K 8..15 / 24..31.  The B operand
// tile (weight rows, contiguous in K) uses the identical loader.
__device__ __forceinline__ v16bf frag_from_lds(const bf16* tile, int row0, int lane) {
  int m   = lane & 15;
  int sel = lane >> 4;
  const bf16* r = tile + (row0 + m) * LDT + sel * 8;
  v8bf lo = *(const v8bf*)(r);
  v8bf hi = *(const v8bf*)(r + 16);
  v16bf f;
#pragma unroll
  for (int i = 0; i < 8; ++i) { f[i] = lo[i]; f[i + 8] = hi[i]; }
  return f;
}

__global__ __launch_bounds__(256)
void gemm_bf16_tn(const bf16* __restrict__ A, const bf16* __restrict__ Bw,
                  const float* __restrict__ bias, float* __restrict__ C,
                  int M, int N, int K) {
#if ASYNC_LDS
  __shared__ __align__(16) bf16 As[2 * TSZ];
  __shared__ __align__(16) bf16 Bs[2 * TSZ];
#else
  __shared__ __align__(16) bf16 As[TSZ];
  __shared__ __align__(16) bf16 Bs[TSZ];
#endif

  const int tid  = threadIdx.x;
  const int lane = tid & 31;
  const int wave = tid >> 5;
  const int wm = wave >> 2;          // 0..1   (64 rows each)
  const int wn = wave & 3;           // 0..3   (32 cols each)
  const int mBase = blockIdx.y * 128;
  const int nBase = blockIdx.x * 128;

  v8f acc[4][2];
  v8f zero8 = {0.f, 0.f, 0.f, 0.f, 0.f, 0.f, 0.f, 0.f};
#pragma unroll
  for (int i = 0; i < 4; ++i) { acc[i][0] = zero8; acc[i][1] = zero8; }

  // cooperative copy of one 128x32 A-tile + B-tile (512 x 16B chunks)
  auto issue_tile = [&](bf16* As_buf, bf16* Bs_buf, int kt) {
#pragma unroll
    for (int i = 0; i < 2; ++i) {
      int chunk = tid + i * 256;
      int r = chunk >> 2;
      int c = (chunk & 3) * 8;
      const bf16* ga = A + (size_t)(mBase + r) * K + kt + c;
      int nr = nBase + r; if (nr >= N) nr = N - 1;   // edge clamp (N tail)
      const bf16* gb = Bw + (size_t)nr * K + kt + c;
#if ASYNC_LDS
      async_copy16(ga, &As_buf[r * LDT + c]);
      async_copy16(gb, &Bs_buf[r * LDT + c]);
#else
      *(uint4*)&As_buf[r * LDT + c] = *(const uint4*)ga;
      *(uint4*)&Bs_buf[r * LDT + c] = *(const uint4*)gb;
#endif
      // speculative prefetch of the tile after next (dropped if OOB)
      __builtin_prefetch(ga + 2 * BK, 0, 1);
      __builtin_prefetch(gb + 2 * BK, 0, 1);
    }
  };

  auto compute_tile = [&](const bf16* As_buf, const bf16* Bs_buf) {
    v16bf bf0 = frag_from_lds(Bs_buf, wn * 32,      lane);
    v16bf bf1 = frag_from_lds(Bs_buf, wn * 32 + 16, lane);
#pragma unroll
    for (int i = 0; i < 4; ++i) {
      v16bf af = frag_from_lds(As_buf, wm * 64 + i * 16, lane);
      acc[i][0] = wmma_bf16(af, bf0, acc[i][0]);
      acc[i][1] = wmma_bf16(af, bf1, acc[i][1]);
    }
  };

#if ASYNC_LDS
  const int nt = K / BK;
  issue_tile(&As[0], &Bs[0], 0);
  for (int t = 0; t < nt; ++t) {
    const int cur = (t & 1) * TSZ;
    if (t + 1 < nt) {
      // other buffer was last read in iteration t-1; the barrier at the end
      // of that iteration makes it safe to overwrite now
      const int nxt = ((t + 1) & 1) * TSZ;
      issue_tile(&As[nxt], &Bs[nxt], (t + 1) * BK);
      wait_async4();                 // tile t's 4 copies done (in-order)
    } else {
      wait_async0();
    }
    __syncthreads();                 // all waves' tile-t data visible
    compute_tile(&As[cur], &Bs[cur]);
    __syncthreads();                 // protect buffer before t+1 overwrites
  }
#else
  for (int kt = 0; kt < K; kt += BK) {
    __syncthreads();
    issue_tile(As, Bs, kt);
    __syncthreads();
    compute_tile(As, Bs);
  }
#endif

  // C/D layout: VGPR r -> lanes 0-15 M=r, lanes 16-31 M=r+8; N = lane&15.
  const int rowSel  = (lane >> 4) * 8;
  const int colBase = nBase + wn * 32 + (lane & 15);
#pragma unroll
  for (int i = 0; i < 4; ++i) {
    int row0 = mBase + wm * 64 + i * 16 + rowSel;
#pragma unroll
    for (int j = 0; j < 2; ++j) {
      int col = colBase + j * 16;
      if (col < N) {
        float bv = bias ? bias[col] : 0.f;
#pragma unroll
        for (int r2 = 0; r2 < 8; ++r2)
          C[(size_t)(row0 + r2) * N + col] = acc[i][j][r2] + bv;
      }
    }
  }
}

// ---------------------------------------------------------------------------
// Small utility kernels
// ---------------------------------------------------------------------------
__global__ void zero2_kernel(float* a) { if (threadIdx.x < 2) a[threadIdx.x] = 0.f; }

__global__ void f32_to_bf16_kernel(const float* __restrict__ src,
                                   bf16* __restrict__ dst, long n) {
  long i = (long)blockIdx.x * blockDim.x + threadIdx.x;
  long stride = (long)gridDim.x * blockDim.x;
  for (; i < n; i += stride) dst[i] = to_bf16(src[i]);
}

// per-column mean/var of scalar_feats [B,16]
__global__ void scalar_stats_kernel(const float* __restrict__ X,
                                    float* __restrict__ mean, float* __restrict__ var) {
  int s = blockIdx.x, tid = threadIdx.x;
  float sm = 0.f, sq = 0.f;
  for (int b = tid; b < BSZ; b += 256) { float v = X[b * 16 + s]; sm += v; sq += v * v; }
  __shared__ float s1[256], s2[256];
  s1[tid] = sm; s2[tid] = sq; __syncthreads();
  for (int o = 128; o > 0; o >>= 1) {
    if (tid < o) { s1[tid] += s1[tid + o]; s2[tid] += s2[tid + o]; }
    __syncthreads();
  }
  if (tid == 0) { float m = s1[0] / BSZ; mean[s] = m; var[s] = s2[0] / BSZ - m * m; }
}

// per-channel stats of z = feats[B,8] @ W[E,8]^T for desc (blocks 0..127) and
// imgs (blocks 128..255)
__global__ void branch_stats_kernel(const float* __restrict__ Fd, const float* __restrict__ Fi,
                                    const float* __restrict__ Wd, const float* __restrict__ Wi,
                                    float* dm, float* dr, float* im, float* ir) {
  int blk = blockIdx.x, tid = threadIdx.x;
  int e = blk & 127;
  const float* F = (blk < 128) ? Fd : Fi;
  const float* W = (blk < 128) ? Wd : Wi;
  float w[8];
#pragma unroll
  for (int k = 0; k < 8; ++k) w[k] = W[e * 8 + k];
  float sm = 0.f, sq = 0.f;
  for (int b = tid; b < BSZ; b += 256) {
    float z = 0.f;
#pragma unroll
    for (int k = 0; k < 8; ++k) z += F[b * 8 + k] * w[k];
    sm += z; sq += z * z;
  }
  __shared__ float s1[256], s2[256];
  s1[tid] = sm; s2[tid] = sq; __syncthreads();
  for (int o = 128; o > 0; o >>= 1) {
    if (tid < o) { s1[tid] += s1[tid + o]; s2[tid] += s2[tid + o]; }
    __syncthreads();
  }
  if (tid == 0) {
    float m = s1[0] / BSZ, v = s2[0] / BSZ - m * m;
    if (blk < 128) { dm[e] = m; dr[e] = rsqrtf(v + EPSV); }
    else           { im[e] = m; ir[e] = rsqrtf(v + EPSV); }
  }
}

// build hcat[B, 2048] (bf16): [5E lag_prod | 5E lag_cat | 5E lag_url | E x]
__global__ void build_hcat_kernel(
    const float* __restrict__ scalar_feats, const float* __restrict__ lag_desc,
    const float* __restrict__ lag_imgs,
    const int* __restrict__ lag_products, const int* __restrict__ first_product,
    const int* __restrict__ lag_categories, const int* __restrict__ first_category,
    const int* __restrict__ lag_urls, const int* __restrict__ first_url,
    const int* __restrict__ cat_feats,
    const float* __restrict__ embP, const float* __restrict__ embC,
    const float* __restrict__ embU, const float* __restrict__ embS,
    const float* __restrict__ W_scalar, const float* __restrict__ g_scalar,
    const float* __restrict__ b_scalar,
    const float* __restrict__ W_desc, const float* __restrict__ g_desc,
    const float* __restrict__ b_desc,
    const float* __restrict__ W_imgs, const float* __restrict__ g_imgs,
    const float* __restrict__ b_imgs,
    const float* __restrict__ scal_mean, const float* __restrict__ scal_var,
    const float* __restrict__ desc_mean, const float* __restrict__ desc_rstd,
    const float* __restrict__ imgs_mean, const float* __restrict__ imgs_rstd,
    bf16* __restrict__ hcat) {
  const int b = blockIdx.x;
  const int e = threadIdx.x;          // 0..127
  const int cat_off[7] = {0, 7, 31, 33, 37, 39, 41};

  // first-event embeddings
  float x = embP[(size_t)first_product[b]  * EDIM + e]
          + embC[(size_t)first_category[b] * EDIM + e]
          + embU[(size_t)first_url[b]      * EDIM + e];
  // 7 small categorical embeddings
#pragma unroll
  for (int j = 0; j < 7; ++j)
    x += embS[(size_t)(cat_off[j] + cat_feats[b * 7 + j]) * EDIM + e];
  // 16 scalar branches: BN stats factorize (y = s*w)
#pragma unroll 4
  for (int s = 0; s < 16; ++s) {
    float w = W_scalar[s * EDIM + e];
    float y = (scalar_feats[b * 16 + s] - scal_mean[s]) * w;
    float t = y * rsqrtf(scal_var[s] * w * w + EPSV)
            * g_scalar[s * EDIM + e] + b_scalar[s * EDIM + e];
    x += fmaxf(t, 0.f);
  }
  // desc branch
  {
    float z = 0.f;
#pragma unroll
    for (int k = 0; k < 8; ++k) z += lag_desc[b * 8 + k] * W_desc[e * 8 + k];
    float t = (z - desc_mean[e]) * desc_rstd[e] * g_desc[e] + b_desc[e];
    x += fmaxf(t, 0.f);
  }
  // imgs branch
  {
    float z = 0.f;
#pragma unroll
    for (int k = 0; k < 8; ++k) z += lag_imgs[b * 8 + k] * W_imgs[e * 8 + k];
    float t = (z - imgs_mean[e]) * imgs_rstd[e] * g_imgs[e] + b_imgs[e];
    x += fmaxf(t, 0.f);
  }

  bf16* row = hcat + (size_t)b * DIN;
#pragma unroll
  for (int l = 0; l < 5; ++l) {
    row[l * EDIM + e]          = to_bf16(embP[(size_t)lag_products[b * 5 + l]   * EDIM + e]);
    row[(5 + l) * EDIM + e]    = to_bf16(embC[(size_t)lag_categories[b * 5 + l] * EDIM + e]);
    row[(10 + l) * EDIM + e]   = to_bf16(embU[(size_t)lag_urls[b * 5 + l]       * EDIM + e]);
  }
  row[15 * EDIM + e] = to_bf16(x);
}

// per-column mean / rsqrt(var+eps) of X[rows,N]
__global__ void column_stats_kernel(const float* __restrict__ X, int rows, int N,
                                    float* __restrict__ mean, float* __restrict__ rstd) {
  int n = blockIdx.x, tid = threadIdx.x;
  float sm = 0.f, sq = 0.f;
  for (int r = tid; r < rows; r += 256) {
    float v = X[(size_t)r * N + n]; sm += v; sq += v * v;
  }
  __shared__ float s1[256], s2[256];
  s1[tid] = sm; s2[tid] = sq; __syncthreads();
  for (int o = 128; o > 0; o >>= 1) {
    if (tid < o) { s1[tid] += s1[tid + o]; s2[tid] += s2[tid + o]; }
    __syncthreads();
  }
  if (tid == 0) {
    float m = s1[0] / rows, v = s2[0] / rows - m * m;
    mean[n] = m; rstd[n] = rsqrtf(v + EPSV);
  }
}

// fused BN + PReLU + bf16 cast
__global__ void bn_prelu_bf16_kernel(const float* __restrict__ X,
                                     const float* __restrict__ mean,
                                     const float* __restrict__ rstd,
                                     const float* __restrict__ g,
                                     const float* __restrict__ be,
                                     const float* __restrict__ p,
                                     bf16* __restrict__ out, int rows, int N) {
  long total = (long)rows * N;
  long i = (long)blockIdx.x * blockDim.x + threadIdx.x;
  long stride = (long)gridDim.x * blockDim.x;
  float slope = p[0];
  for (; i < total; i += stride) {
    int n = (int)(i % N);
    float v = (X[i] - mean[n]) * rstd[n] * g[n] + be[n];
    out[i] = to_bf16(v > 0.f ? v : slope * v);
  }
}

// per-row online logsumexp over V logits + masked NLL accumulation
__global__ void row_loss_kernel(const float* __restrict__ logits,
                                const int* __restrict__ target,
                                float* __restrict__ accum, int N) {
  int b = blockIdx.x, tid = threadIdx.x;
  const float* row = logits + (size_t)b * N;
  float m = -1e30f, s = 0.f;
  for (int i = tid; i < N; i += 256) {
    float v = row[i];
    if (v > m) { s = s * __expf(m - v) + 1.f; m = v; }
    else       { s += __expf(v - m); }
  }
  __shared__ float shm[256], shs[256];
  shm[tid] = m; shs[tid] = s; __syncthreads();
  for (int o = 128; o > 0; o >>= 1) {
    if (tid < o) {
      float m2 = shm[tid + o], s2 = shs[tid + o];
      float M = fmaxf(shm[tid], m2);
      shs[tid] = shs[tid] * __expf(shm[tid] - M) + s2 * __expf(m2 - M);
      shm[tid] = M;
    }
    __syncthreads();
  }
  if (tid == 0) {
    int t = target[b];
    if (t != 0) {                         // ignore_index == 0
      float nll = (shm[0] + __logf(shs[0])) - row[t];
      atomicAdd(&accum[0], nll);
      atomicAdd(&accum[1], 1.f);
    }
  }
}

__global__ void finalize_loss_kernel(const float* __restrict__ accum,
                                     float* __restrict__ out_loss) {
  if (threadIdx.x == 0 && blockIdx.x == 0) out_loss[0] = accum[0] / accum[1];
}

// ---------------------------------------------------------------------------
// Host launcher
// ---------------------------------------------------------------------------
extern "C" void kernel_launch(void* const* d_in, const int* in_sizes, int n_in,
                              void* d_out, int out_size, void* d_ws, size_t ws_size,
                              hipStream_t stream) {
  const float* scalar_feats  = (const float*)d_in[0];
  const float* lag_desc      = (const float*)d_in[1];
  const float* lag_imgs      = (const float*)d_in[2];
  const int*   lag_products  = (const int*)d_in[3];
  const int*   first_product = (const int*)d_in[4];
  const int*   lag_categories= (const int*)d_in[5];
  const int*   first_category= (const int*)d_in[6];
  const int*   lag_urls      = (const int*)d_in[7];
  const int*   first_url     = (const int*)d_in[8];
  const int*   cat_feats     = (const int*)d_in[9];
  const int*   target        = (const int*)d_in[10];
  const float* embP          = (const float*)d_in[11];
  const float* embC          = (const float*)d_in[12];
  const float* embU          = (const float*)d_in[13];
  const float* embS          = (const float*)d_in[14];
  const float* W_scalar      = (const float*)d_in[15];
  const float* g_scalar      = (const float*)d_in[16];
  const float* b_scalar      = (const float*)d_in[17];
  const float* W_desc        = (const float*)d_in[18];
  const float* g_desc        = (const float*)d_in[19];
  const float* b_desc        = (const float*)d_in[20];
  const float* W_imgs        = (const float*)d_in[21];
  const float* g_imgs        = (const float*)d_in[22];
  const float* b_imgs        = (const float*)d_in[23];
  const float* W1 = (const float*)d_in[24];
  const float* g1 = (const float*)d_in[26];
  const float* be1= (const float*)d_in[27];
  const float* p1 = (const float*)d_in[28];
  const float* W2 = (const float*)d_in[29];
  const float* g2 = (const float*)d_in[31];
  const float* be2= (const float*)d_in[32];
  const float* p2 = (const float*)d_in[33];
  const float* W3 = (const float*)d_in[34];
  const float* g3 = (const float*)d_in[36];
  const float* be3= (const float*)d_in[37];
  const float* p3 = (const float*)d_in[38];
  const float* out_bias = (const float*)d_in[39];
  // b1/b2/b3 (d_in[25],[30],[35]) cancel under training-mode BN -> skipped.

  float* logits = (float*)d_out;
  float* loss   = logits + (size_t)BSZ * VOCAB;

  // workspace carve (256B aligned)
  char* base = (char*)d_ws;
  size_t off = 0;
  auto take = [&](size_t nbytes) -> void* {
    off = (off + 255) & ~(size_t)255;
    void* r = base + off;
    off += nbytes;
    return r;
  };
  float* scal_mean = (float*)take(16 * 4);
  float* scal_var  = (float*)take(16 * 4);
  float* desc_mean = (float*)take(EDIM * 4);
  float* desc_rstd = (float*)take(EDIM * 4);
  float* imgs_mean = (float*)take(EDIM * 4);
  float* imgs_rstd = (float*)take(EDIM * 4);
  float* bn_mean   = (float*)take(HDIM * 4);     // reused for all 3 BN layers
  float* bn_rstd   = (float*)take(HDIM * 4);
  float* accum     = (float*)take(2 * 4);
  bf16*  hcatb = (bf16*)take((size_t)BSZ * DIN * 2);
  bf16*  W1b   = (bf16*)take((size_t)HDIM * DIN * 2);
  bf16*  W2b   = (bf16*)take((size_t)HDIM * HDIM * 2);
  bf16*  W3b   = (bf16*)take((size_t)EDIM * HDIM * 2);
  bf16*  embPb = (bf16*)take((size_t)VOCAB * EDIM * 2);
  float* hf    = (float*)take((size_t)BSZ * HDIM * 4);   // h1 -> h2 -> h3 (reused)
  bf16*  hb    = (bf16*)take((size_t)BSZ * HDIM * 2);    // h1b -> h2b (reused)
  bf16*  h3b   = (bf16*)take((size_t)BSZ * EDIM * 2);

  zero2_kernel<<<1, 32, 0, stream>>>(accum);
  scalar_stats_kernel<<<16, 256, 0, stream>>>(scalar_feats, scal_mean, scal_var);
  branch_stats_kernel<<<256, 256, 0, stream>>>(lag_desc, lag_imgs, W_desc, W_imgs,
                                               desc_mean, desc_rstd, imgs_mean, imgs_rstd);
  build_hcat_kernel<<<BSZ, EDIM, 0, stream>>>(
      scalar_feats, lag_desc, lag_imgs, lag_products, first_product,
      lag_categories, first_category, lag_urls, first_url, cat_feats,
      embP, embC, embU, embS, W_scalar, g_scalar, b_scalar,
      W_desc, g_desc, b_desc, W_imgs, g_imgs, b_imgs,
      scal_mean, scal_var, desc_mean, desc_rstd, imgs_mean, imgs_rstd, hcatb);

  f32_to_bf16_kernel<<<2048, 256, 0, stream>>>(W1, W1b, (long)HDIM * DIN);
  f32_to_bf16_kernel<<<1024, 256, 0, stream>>>(W2, W2b, (long)HDIM * HDIM);
  f32_to_bf16_kernel<<<128, 256, 0, stream>>>(W3, W3b, (long)EDIM * HDIM);
  f32_to_bf16_kernel<<<8192, 256, 0, stream>>>(embP, embPb, (long)VOCAB * EDIM);

  // MLP layer 1: [2048,2048] @ [1024,2048]^T
  gemm_bf16_tn<<<dim3(HDIM / 128, BSZ / 128), 256, 0, stream>>>(
      hcatb, W1b, nullptr, hf, BSZ, HDIM, DIN);
  column_stats_kernel<<<HDIM, 256, 0, stream>>>(hf, BSZ, HDIM, bn_mean, bn_rstd);
  bn_prelu_bf16_kernel<<<4096, 256, 0, stream>>>(hf, bn_mean, bn_rstd, g1, be1, p1,
                                                 hb, BSZ, HDIM);
  // MLP layer 2: [2048,1024] @ [1024,1024]^T
  gemm_bf16_tn<<<dim3(HDIM / 128, BSZ / 128), 256, 0, stream>>>(
      hb, W2b, nullptr, hf, BSZ, HDIM, HDIM);
  column_stats_kernel<<<HDIM, 256, 0, stream>>>(hf, BSZ, HDIM, bn_mean, bn_rstd);
  bn_prelu_bf16_kernel<<<4096, 256, 0, stream>>>(hf, bn_mean, bn_rstd, g2, be2, p2,
                                                 hb, BSZ, HDIM);
  // MLP layer 3: [2048,1024] @ [128,1024]^T
  gemm_bf16_tn<<<dim3(EDIM / 128, BSZ / 128), 256, 0, stream>>>(
      hb, W3b, nullptr, hf, BSZ, EDIM, HDIM);
  column_stats_kernel<<<EDIM, 256, 0, stream>>>(hf, BSZ, EDIM, bn_mean, bn_rstd);
  bn_prelu_bf16_kernel<<<512, 256, 0, stream>>>(hf, bn_mean, bn_rstd, g3, be3, p3,
                                                h3b, BSZ, EDIM);
  // tied-weight output layer: [2048,128] @ [100000,128]^T + out_bias
  gemm_bf16_tn<<<dim3((VOCAB + 127) / 128, BSZ / 128), 256, 0, stream>>>(
      h3b, embPb, out_bias, logits, BSZ, VOCAB, EDIM);

  row_loss_kernel<<<BSZ, 256, 0, stream>>>(logits, target, accum, VOCAB);
  finalize_loss_kernel<<<1, 32, 0, stream>>>(accum, loss);
}
